// Encoder_91122026151950
// MI455X (gfx1250) — matrix operand order repaired and checked by reading, not compiled, not run on hardware
//
#include <hip/hip_runtime.h>
#include <stdint.h>

// ---------------- problem constants ----------------
#define SEQ     240
#define HID     180
#define BATCH   2048
#define NROWS   (BATCH * SEQ)        // 491520 independent rows
#define NTILES  (NROWS / 16)         // 30720 M-tiles of 16 rows
#define KT      6                    // K padded 180 -> 192 = 6 steps of 32 (f16 wmma)
#define JT      12                   // j padded 180 -> 192 = 12 tiles of 16
#define NGS     3                    // gate groups actually needed: i, g, o (f is dead)
#define NT      (NGS * JT)           // 36 B N-tiles resident in LDS
#define FRAG_BYTES 1024              // one 32x16 f16 B fragment (32 lanes * 32B)
#define WFRAG_BYTES (NT * KT * FRAG_BYTES)   // 221184 B
#define CHUNK_ROWS 30
#define NCHUNKS    24                // 24 * 30 = 720 weight rows
#define BIAS_OFF   WFRAG_BYTES
#define BIAS_BYTES (NGS * 192 * 4)   // 2304 B pre-summed biases
#define DYN_OFF    (BIAS_OFF + BIAS_BYTES)       // 223488
#define STAGE_OFF  DYN_OFF                       // phase-1 TDM staging (21600 B)
#define HROW_STRIDE 196                          // padded dwords per h row (bank-conflict free)
#define HBUF_STRIDE (HROW_STRIDE * 16 * 4 + 64)  // 12608 B per wave (h tile + 16 inv values)
#define HINV_OFF   (HROW_STRIDE * 16 * 4)        // 12544
#define SMEM_BYTES (DYN_OFF + 8 * HBUF_STRIDE)   // 324352 <= 320KB WGP LDS

#define GRID_BLOCKS 1024
#define BLOCK_THREADS 256            // 8 wave32 per block

typedef __attribute__((ext_vector_type(16))) _Float16 v16h;
typedef __attribute__((ext_vector_type(8)))  float    v8f;
typedef __attribute__((ext_vector_type(8)))  uint32_t v8u;
typedef __attribute__((ext_vector_type(4)))  float    f32x4;
typedef __attribute__((ext_vector_type(4)))  uint32_t u32x4;
typedef __attribute__((ext_vector_type(8)))  int      i32x8;
typedef __attribute__((ext_vector_type(4)))  int      i32x4;

// ---------------- fast math helpers ----------------
__device__ __forceinline__ float fast_exp(float x) {
#if __has_builtin(__builtin_amdgcn_exp2f)
    return __builtin_amdgcn_exp2f(x * 1.44269504088896340736f);
#else
    return __expf(x);
#endif
}
__device__ __forceinline__ float fast_rcp(float x) {
#if __has_builtin(__builtin_amdgcn_rcpf)
    return __builtin_amdgcn_rcpf(x);
#else
    return 1.0f / x;
#endif
}
// gfx1250 has v_tanh_f32; sigmoid(x) = 0.5 + 0.5*tanh(x/2)
__device__ __forceinline__ float tanh_fast(float x) {
#if __has_builtin(__builtin_amdgcn_tanhf)
    return __builtin_amdgcn_tanhf(x);
#else
    return 2.0f * fast_rcp(1.0f + fast_exp(-2.0f * x)) - 1.0f;
#endif
}
__device__ __forceinline__ float sigmoid_fast(float x) { return 0.5f + 0.5f * tanh_fast(0.5f * x); }

__device__ __forceinline__ uint32_t pack2h(float a, float b) {
    _Float16 ha = (_Float16)a, hb = (_Float16)b;
    uint16_t ua = __builtin_bit_cast(uint16_t, ha);
    uint16_t ub = __builtin_bit_cast(uint16_t, hb);
    return (uint32_t)ua | ((uint32_t)ub << 16);
}

// xs[row, i] gather: rowp already points at b*43200 + (s%80)*540 + s/80
__device__ __forceinline__ float load_x(const float* rowp, int i) {
    int ic = (i < HID) ? i : 0;
    unsigned q = (unsigned)ic / 60u;
    unsigned r = (unsigned)ic - q * 60u;
    float v = rowp[q * 180u + r * 3u];
    return (i < HID) ? v : 0.0f;
}

__global__ __launch_bounds__(BLOCK_THREADS)
void Encoder_91122026151950_kernel(const float* __restrict__ x,
                                   const float* __restrict__ W_ih,
                                   const float* __restrict__ b_ih,
                                   const float* __restrict__ b_hh,
                                   float* __restrict__ out) {
    extern __shared__ char smem[];
    char*  wfrag = smem;                               // 216 KB f16 B fragments
    float* biasb = (float*)(smem + BIAS_OFF);          // pre-summed biases [3][192]
    float* stage = (float*)(smem + STAGE_OFF);         // phase-1 f32 TDM staging

    const int tid    = threadIdx.x;
    const int lane   = tid & 31;
    const int waveId = tid >> 5;

    // ============ Phase 1a: pre-sum biases into LDS ============
    for (int i = tid; i < NGS * 192; i += BLOCK_THREADS) {
        int gsel = i / 192, jp = i - gsel * 192;
        int gr = (gsel == 0) ? 0 : gsel + 1;           // 0->i, 1->g, 2->o
        biasb[i] = (jp < HID) ? (b_ih[gr * HID + jp] + b_hh[gr * HID + jp]) : 0.0f;
    }

    // ============ Phase 1b: stage W (3 live gate groups) as f16 WMMA B fragments ============
    for (int c = 0; c < NCHUNKS; ++c) {
        if (waveId == 0) {
#if __has_builtin(__builtin_amdgcn_tensor_load_to_lds)
            // Tensor Data Mover: 2D tile [30 rows x 180 f32] global -> LDS staging
            uint64_t ga  = (uint64_t)(uintptr_t)(W_ih + (size_t)c * CHUNK_ROWS * HID);
            uint32_t lds = (uint32_t)(uintptr_t)stage;   // low 32 bits = LDS offset
            u32x4 g0 = { 1u,                                    // count=1, gather off
                         lds,                                   // lds_addr
                         (uint32_t)ga,                          // global_addr[31:0]
                         ((uint32_t)(ga >> 32) & 0x01FFFFFFu) | (2u << 30) }; // addr[56:32] | type=2
            i32x8 g1 = { 0x20000,               // data_size=2 (4B), no mask/pad/iterate
                         (int)(180u << 16),     // tensor_dim0[15:0]
                         (int)( 30u << 16),     // tensor_dim1[15:0]
                         (int)(180u << 16),     // tile_dim0
                         30,                    // tile_dim1
                         180,                   // tensor_dim0_stride
                         0, 0 };
            i32x4 gz = { 0, 0, 0, 0 };
#if defined(__clang_major__) && (__clang_major__ >= 23)
            i32x8 gz8 = { 0, 0, 0, 0, 0, 0, 0, 0 };
            __builtin_amdgcn_tensor_load_to_lds(g0, g1, gz, gz, gz8, 0);
#else
            __builtin_amdgcn_tensor_load_to_lds(g0, g1, gz, gz, 0);
#endif
            __builtin_amdgcn_s_wait_tensorcnt(0);
#else
            for (int idx = lane; idx < CHUNK_ROWS * HID; idx += 32)
                stage[idx] = W_ih[(size_t)c * CHUNK_ROWS * HID + idx];
#endif
        }
        __syncthreads();

        // convert this chunk of weight rows into fragment layout
        for (int task = tid; task < CHUNK_ROWS * 12; task += BLOCK_THREADS) {
            int r  = task / 12;          // row in chunk
            int kg = task % 12;          // K-group of 16
            int g  = c * CHUNK_ROWS + r; // gate row 0..719
            int gr = g / HID;
            if (gr == 1) continue;       // forget gate is dead (c_prev == 0)
            int gsel  = (gr == 0) ? 0 : gr - 1;
            int jp    = g - gr * HID;
            int jt    = jp >> 4;
            int lfrag = (jp & 15) + ((kg & 1) << 4);   // B layout: N=lane&15, K-half by lane>=16
            int t     = kg >> 1;
            int kbase = t * 32 + (kg & 1) * 16;
            const float* src = stage + r * HID + kbase;
            v8u pk;
            #pragma unroll
            for (int e2 = 0; e2 < 8; ++e2) {
                int k0 = kbase + 2 * e2;
                float a = (k0     < HID) ? src[2 * e2]     : 0.0f;
                float b = (k0 + 1 < HID) ? src[2 * e2 + 1] : 0.0f;
                pk[e2] = pack2h(a, b);
            }
            *(v8u*)(wfrag + ((gsel * JT + jt) * KT + t) * FRAG_BYTES + lfrag * 32) = pk;
        }
        __syncthreads();   // last barrier also publishes biasb; stage buffer dead after loop
    }

    // ============ Phase 2: persistent waves over 16-row M tiles ============
    float* hb   = (float*)(smem + DYN_OFF + waveId * HBUF_STRIDE);  // per-wave [16][196] f32
    float* hinv = (float*)((char*)hb + HINV_OFF);                   // per-wave 16 x inv(sum)

    const int gwave  = blockIdx.x * (BLOCK_THREADS >> 5) + waveId;
    const int nWaves = GRID_BLOCKS * (BLOCK_THREADS >> 5);
    const int khalf  = (lane & 16);        // A fragment K offset: 0 / 16
    const int lanelo = lane & 15;
    const int rOff   = (lane & 16) ? 8 : 0;
    const bool v11   = lanelo < 4;         // valid lanes of the padded last j-tile

    #pragma unroll 1
    for (int tile = gwave; tile < NTILES; tile += nWaves) {
        int myRow = tile * 16 + lanelo;
        int b = myRow / SEQ;
        int s = myRow - b * SEQ;
        const float* rowp = x + (size_t)b * (SEQ * HID) + (s % 80) * 540 + (s / 80);

        // prefetch next tile's rows (global_prefetch_b8)
        int ntile = tile + nWaves;
        if (ntile < NTILES) {
            int nr = ntile * 16 + lanelo;
            int nb = nr / SEQ, ns = nr - nb * SEQ;
            __builtin_prefetch(x + (size_t)nb * (SEQ * HID) + (ns % 80) * 540 + (ns / 80), 0, 0);
        }

        // --- gather A fragments (16x32 f16 per k-step) straight into registers ---
        v16h A[KT];
        #pragma unroll
        for (int t = 0; t < KT; ++t) {
            v8u pk;
            #pragma unroll
            for (int r = 0; r < 8; ++r) {
                int e  = 2 * r;
                int k0 = t * 32 + khalf + e;
                pk[r] = pack2h(load_x(rowp, k0), load_x(rowp, k0 + 1));
            }
            A[t] = __builtin_bit_cast(v16h, pk);
        }

        // --- GEMM + fused LSTM activation; h tile goes to LDS, running max in regs ---
        v8f m;
        #pragma unroll
        for (int q = 0; q < 8; ++q) m[q] = -3.0e38f;

        #pragma unroll 1                       // keep register pressure bounded
        for (int jt = 0; jt < JT; ++jt) {
            int jp = jt * 16 + lanelo;
            v8f a0, a1, a2;
            {
                float b0 = biasb[0 * 192 + jp];
                float b1 = biasb[1 * 192 + jp];
                float b2 = biasb[2 * 192 + jp];
                #pragma unroll
                for (int q = 0; q < 8; ++q) { a0[q] = b0; a1[q] = b1; a2[q] = b2; }
            }
            const char* fb = wfrag + jt * (KT * FRAG_BYTES) + lane * 32;
            #pragma unroll
            for (int t = 0; t < KT; ++t) {
                v16h B0 = *(const v16h*)(fb + 0 * (JT * KT * FRAG_BYTES) + t * FRAG_BYTES);
                v16h B1 = *(const v16h*)(fb + 1 * (JT * KT * FRAG_BYTES) + t * FRAG_BYTES);
                v16h B2 = *(const v16h*)(fb + 2 * (JT * KT * FRAG_BYTES) + t * FRAG_BYTES);
                a0 = __builtin_amdgcn_wmma_f32_16x16x32_f16(false, A[t], false, B0, (short)0, a0, false, false);
                a1 = __builtin_amdgcn_wmma_f32_16x16x32_f16(false, A[t], false, B1, (short)0, a1, false, false);
                a2 = __builtin_amdgcn_wmma_f32_16x16x32_f16(false, A[t], false, B2, (short)0, a2, false, false);
            }
            bool live = (jt != JT - 1) || v11;
            float* hcol = hb + rOff * HROW_STRIDE + jt * 16 + lanelo;
            #pragma unroll
            for (int q = 0; q < 8; ++q) {
                float ig = sigmoid_fast(a0[q]);
                float gg = tanh_fast(a1[q]);
                float og = sigmoid_fast(a2[q]);
                float hv = og * tanh_fast(ig * gg);
                hcol[q * HROW_STRIDE] = hv;                    // ds_store_b32, conflict-free
                m[q] = fmaxf(m[q], live ? hv : -3.0e38f);
            }
        }

        // --- softmax over 180 j values ---
        #pragma unroll
        for (int mk = 1; mk <= 8; mk <<= 1) {
            #pragma unroll
            for (int q = 0; q < 8; ++q) m[q] = fmaxf(m[q], __shfl_xor(m[q], mk, 32));
        }
        v8f sum;
        #pragma unroll
        for (int q = 0; q < 8; ++q) sum[q] = 0.0f;
        #pragma unroll 1
        for (int jt = 0; jt < JT; ++jt) {
            bool live = (jt != JT - 1) || v11;
            float* hcol = hb + rOff * HROW_STRIDE + jt * 16 + lanelo;
            #pragma unroll
            for (int q = 0; q < 8; ++q) {
                float e = fast_exp(hcol[q * HROW_STRIDE] - m[q]);
                e = live ? e : 0.0f;
                hcol[q * HROW_STRIDE] = e;
                sum[q] += e;
            }
        }
        #pragma unroll
        for (int mk = 1; mk <= 8; mk <<= 1) {
            #pragma unroll
            for (int q = 0; q < 8; ++q) sum[q] += __shfl_xor(sum[q], mk, 32);
        }
        if (lanelo == 0) {   // lane 0 -> rows 0..7, lane 16 -> rows 8..15
            #pragma unroll
            for (int q = 0; q < 8; ++q) hinv[q + rOff] = fast_rcp(sum[q]);
        }
        asm volatile("s_wait_dscnt 0x0" ::: "memory");   // in-wave LDS handoff

        // --- coalesced scaled store: 45 float4 per row ---
        size_t obase = (size_t)tile * 16 * HID;
        #pragma unroll 1
        for (int i = lane; i < 16 * 45; i += 32) {
            unsigned row = (unsigned)i / 45u;
            unsigned cc  = (unsigned)i - row * 45u;
            f32x4 v = *(const f32x4*)(hb + row * HROW_STRIDE + cc * 4);
            float iv = hinv[row];
            v.x *= iv; v.y *= iv; v.z *= iv; v.w *= iv;
            *(f32x4*)(out + obase + (size_t)row * HID + cc * 4) = v;   // global_store_b128
        }
    }
}

extern "C" void kernel_launch(void* const* d_in, const int* in_sizes, int n_in,
                              void* d_out, int out_size, void* d_ws, size_t ws_size,
                              hipStream_t stream) {
    (void)in_sizes; (void)n_in; (void)out_size; (void)d_ws; (void)ws_size;
    const float* x    = (const float*)d_in[0];
    const float* W_ih = (const float*)d_in[1];
    // d_in[2] = W_hh: multiplies the all-zero state, contributes nothing
    const float* b_ih = (const float*)d_in[3];
    const float* b_hh = (const float*)d_in[4];
    float* out = (float*)d_out;

    hipFuncSetAttribute((const void*)Encoder_91122026151950_kernel,
                        hipFuncAttributeMaxDynamicSharedMemorySize, SMEM_BYTES);
    Encoder_91122026151950_kernel<<<GRID_BLOCKS, BLOCK_THREADS, SMEM_BYTES, stream>>>(
        x, W_ih, b_ih, b_hh, out);
}
